// CrossAttentionModule_76184129896947
// MI455X (gfx1250) — matrix operand order repaired and checked by reading, compiled, not run on hardware
//
#include <hip/hip_runtime.h>
#include <hip/hip_bf16.h>

// ---------------------------------------------------------------------------
// Types for CDNA5 WMMA (wave32, v_wmma_f32_16x16x32_bf16)
// ---------------------------------------------------------------------------
typedef __attribute__((ext_vector_type(16))) __bf16 v16bf;
typedef __attribute__((ext_vector_type(8)))  __bf16 v8bf;
typedef __attribute__((ext_vector_type(4)))  __bf16 v4bf;
typedef __attribute__((ext_vector_type(8)))  float  v8f;

__device__ __forceinline__ v16bf pack16(v8bf lo, v8bf hi) {
    v16bf r;
#pragma unroll
    for (int i = 0; i < 8; ++i) { r[i] = lo[i]; r[8 + i] = hi[i]; }
    return r;
}

__device__ __forceinline__ v4bf cvt4(float4 f) {
    v4bf r;
    r[0] = (__bf16)f.x; r[1] = (__bf16)f.y;
    r[2] = (__bf16)f.z; r[3] = (__bf16)f.w;
    return r;
}

__device__ __forceinline__ v8f wmma_bf16(v16bf a, v16bf b, v8f c) {
    return __builtin_amdgcn_wmma_f32_16x16x32_bf16(
        /*neg_a=*/false, a, /*neg_b=*/false, b,
        /*c_mod=*/(short)0, c, /*reuse_a=*/false, /*reuse_b=*/false);
}

// ---------------------------------------------------------------------------
// Generic batched GEMM:  Out[b] = A(MxK) @ X[b](KxN) (+bias) (+residual)
// Block = 128 threads (4 waves), tile 64(M) x 64(N), K-step 32, bf16 WMMA.
// Requires M%64==0 (via grid.y), N%64==0, K%32==0. All call sites comply.
// ---------------------------------------------------------------------------
#define LDA 40  // padded LDS row stride in bf16 elements (80B, 16B-aligned)

__global__ __launch_bounds__(128, 1)
void gemm64x64_bf16wmma(const float* __restrict__ A,
                        const float* __restrict__ X,
                        const float* __restrict__ bias,
                        const float* __restrict__ R,
                        float* __restrict__ O,
                        int M, int N, int K)
{
    __shared__ __bf16 As[64][LDA];
    __shared__ __bf16 Bt[64][LDA];   // B tile stored transposed: Bt[n][k]

    const int b    = blockIdx.z;
    const int m0   = blockIdx.y * 64;
    const int n0   = blockIdx.x * 64;
    const float* Xb = X + (size_t)b * K * N;

    const int tid  = threadIdx.x;
    const int lane = tid & 31;
    const int wave = tid >> 5;
    const int wm   = (wave >> 1) * 32;   // wave's M offset in the 64-tile
    const int wn   = (wave & 1) * 32;    // wave's N offset in the 64-tile
    const int half = lane >> 4;
    const int l16  = lane & 15;

    v8f acc[2][2];
#pragma unroll
    for (int i = 0; i < 2; ++i)
#pragma unroll
        for (int j = 0; j < 2; ++j)
            acc[i][j] = (v8f){0.f, 0.f, 0.f, 0.f, 0.f, 0.f, 0.f, 0.f};

    for (int k0 = 0; k0 < K; k0 += 32) {
        // ---- batched wide staging: 8 global_load_b128 in flight per thread ----
        float4 abuf[4], bbuf[4];
#pragma unroll
        for (int j = 0; j < 4; ++j) {
            int idx = tid + j * 128;             // [0,512)
            int ar  = idx >> 3, ac4 = (idx & 7) * 4;     // A: 64 rows x 8 col4
            abuf[j] = *(const float4*)&A[(size_t)(m0 + ar) * K + (k0 + ac4)];
            int bk  = idx >> 4, bn4 = (idx & 15) * 4;    // B: 32 k x 16 n4
            bbuf[j] = *(const float4*)&Xb[(size_t)(k0 + bk) * N + (n0 + bn4)];
        }
#pragma unroll
        for (int j = 0; j < 4; ++j) {
            int idx = tid + j * 128;
            int ar  = idx >> 3, ac4 = (idx & 7) * 4;
            *(v4bf*)&As[ar][ac4] = cvt4(abuf[j]);        // packed ds_store_b64
            int bk  = idx >> 4, bn4 = (idx & 15) * 4;
            Bt[bn4 + 0][bk] = (__bf16)bbuf[j].x;         // transposed scalar
            Bt[bn4 + 1][bk] = (__bf16)bbuf[j].y;
            Bt[bn4 + 2][bk] = (__bf16)bbuf[j].z;
            Bt[bn4 + 3][bk] = (__bf16)bbuf[j].w;
        }
        // prefetch next K tile into cache hierarchy (global_prefetch_b8)
        if (k0 + 32 < K) {
            __builtin_prefetch(&A[(size_t)(m0 + (tid >> 1)) * K + (k0 + 32)], 0, 3);
            __builtin_prefetch(&Xb[(size_t)(k0 + 32 + (tid >> 2)) * N + n0], 0, 3);
        }
        __syncthreads();

        // ---- WMMA: each wave does 2x2 fragments of 16x16 over K=32 ----
#pragma unroll
        for (int mi = 0; mi < 2; ++mi) {
            const int ar = wm + mi * 16 + l16;       // A-matrix row
            const int kb = half * 8;                 // lane-half K base
            v16bf a = pack16(*(const v8bf*)&As[ar][kb],
                             *(const v8bf*)&As[ar][kb + 16]);
#pragma unroll
            for (int ni = 0; ni < 2; ++ni) {
                const int bc = wn + ni * 16 + l16;   // B-matrix column
                const int kh = half * 8;
                v16bf bb = pack16(*(const v8bf*)&Bt[bc][kh],
                                  *(const v8bf*)&Bt[bc][kh + 16]);
                acc[mi][ni] = wmma_bf16(a, bb, acc[mi][ni]);
            }
        }
        __syncthreads();
    }

    // ---- epilogue: C/D layout row = half*8 + r, col = l16 ----
#pragma unroll
    for (int mi = 0; mi < 2; ++mi)
#pragma unroll
        for (int ni = 0; ni < 2; ++ni)
#pragma unroll
            for (int r = 0; r < 8; ++r) {
                int row = m0 + wm + mi * 16 + half * 8 + r;
                int col = n0 + wn + ni * 16 + l16;
                float v = acc[mi][ni][r];
                if (bias) v += bias[row];
                size_t o = (size_t)b * M * N + (size_t)row * N + col;
                if (R) v += R[o];
                O[o] = v;
            }
}

// ---------------------------------------------------------------------------
// Flash attention over channel dim C=256:
//   S = Q^T K * scale ; P = online-softmax(S) ; AV = V @ P^T  (out [C,N])
// Q,K,V,Oav: [B, 256, N] fp32. Block = 128 threads (4 waves x 16 queries),
// streams keys in tiles of 32. Q fragments live in registers all kernel.
// LDS overlay: phase 1 uses Qt[64][264]; the same arena is then reused for
// Kt[32][264] + Vl[256][40] + Pl[4][16][40]  (42,496 B total).
// ---------------------------------------------------------------------------
#define SMEM_KT   0
#define SMEM_VL   16896                 // 32*264*2
#define SMEM_PL   (16896 + 20480)       // + 256*40*2
#define SMEM_SZ   (16896 + 20480 + 5120)

__global__ __launch_bounds__(128, 1)
void attn_flash_bf16wmma(const float* __restrict__ Q,
                         const float* __restrict__ Kx,
                         const float* __restrict__ Vx,
                         float* __restrict__ Oav,
                         int N, float scale)
{
    const int C = 256;
    __shared__ __align__(16) char smem[SMEM_SZ];
    __bf16 (*Qt)[264]     = reinterpret_cast<__bf16(*)[264]>(smem);     // [64][264]
    __bf16 (*Kt)[264]     = reinterpret_cast<__bf16(*)[264]>(smem + SMEM_KT); // [32][264]
    __bf16 (*Vl)[40]      = reinterpret_cast<__bf16(*)[40]>(smem + SMEM_VL);  // [256][40]
    __bf16 (*Pl)[16][40]  = reinterpret_cast<__bf16(*)[16][40]>(smem + SMEM_PL); // [4][16][40]

    const int b   = blockIdx.z;
    const int n0  = blockIdx.x * 64;
    const float* Qb = Q  + (size_t)b * C * N;
    const float* Kb = Kx + (size_t)b * C * N;
    const float* Vb = Vx + (size_t)b * C * N;

    const int tid  = threadIdx.x;
    const int lane = tid & 31;
    const int wave = tid >> 5;
    const int half = lane >> 4;
    const int l16  = lane & 15;

    // ---- phase 1: stage Q tile [64 q][256 c] coalesced, grab fragments ----
    for (int g = 0; g < 8; ++g) {
        float4 qb4[4];
#pragma unroll
        for (int j = 0; j < 4; ++j) {
            int idx = tid + (g * 4 + j) * 128;   // [0,4096)
            int n4 = (idx & 15) * 4, c = idx >> 4;
            qb4[j] = *(const float4*)&Qb[(size_t)c * N + (n0 + n4)];
        }
#pragma unroll
        for (int j = 0; j < 4; ++j) {
            int idx = tid + (g * 4 + j) * 128;
            int n4 = (idx & 15) * 4, c = idx >> 4;
            Qt[n4 + 0][c] = (__bf16)qb4[j].x;
            Qt[n4 + 1][c] = (__bf16)qb4[j].y;
            Qt[n4 + 2][c] = (__bf16)qb4[j].z;
            Qt[n4 + 3][c] = (__bf16)qb4[j].w;
        }
    }
    __syncthreads();
    v16bf qf[8];
    {
        const int qrow = wave * 16 + l16;        // A-fragment row (query)
#pragma unroll
        for (int kc = 0; kc < 8; ++kc) {
            const int c0 = kc * 32 + half * 8;   // lane-half K base
            qf[kc] = pack16(*(const v8bf*)&Qt[qrow][c0],
                            *(const v8bf*)&Qt[qrow][c0 + 16]);
        }
    }
    __syncthreads();   // Qt dead; arena reused for Kt/Vl/Pl below

    // ---- running softmax state (rows = half*8 + r) and O accumulators ----
    float run_m[8], run_l[8];
#pragma unroll
    for (int r = 0; r < 8; ++r) { run_m[r] = -3.0e38f; run_l[r] = 0.f; }
    v8f o[16];
#pragma unroll
    for (int cf = 0; cf < 16; ++cf)
        o[cf] = (v8f){0.f, 0.f, 0.f, 0.f, 0.f, 0.f, 0.f, 0.f};

    for (int m0 = 0; m0 < N; m0 += 32) {
        // ---- stage K and V tiles (32 keys x 256 ch), wide + batched ----
        for (int g = 0; g < 4; ++g) {
            float4 kb4[4], vb4[4];
#pragma unroll
            for (int j = 0; j < 4; ++j) {
                int idx = tid + (g * 4 + j) * 128;   // [0,2048)
                int m4 = (idx & 7) * 4, c = idx >> 3;
                size_t gidx = (size_t)c * N + (m0 + m4);
                kb4[j] = *(const float4*)&Kb[gidx];
                vb4[j] = *(const float4*)&Vb[gidx];
            }
#pragma unroll
            for (int j = 0; j < 4; ++j) {
                int idx = tid + (g * 4 + j) * 128;
                int m4 = (idx & 7) * 4, c = idx >> 3;
                Kt[m4 + 0][c] = (__bf16)kb4[j].x;    // transposed scalar
                Kt[m4 + 1][c] = (__bf16)kb4[j].y;
                Kt[m4 + 2][c] = (__bf16)kb4[j].z;
                Kt[m4 + 3][c] = (__bf16)kb4[j].w;
                *(v4bf*)&Vl[c][m4] = cvt4(vb4[j]);   // packed ds_store_b64
            }
        }
        __syncthreads();

        // ---- S = Q^T K : two 16x16 fragments (key cols 0-15 / 16-31) ----
        v8f s0 = (v8f){0.f,0.f,0.f,0.f,0.f,0.f,0.f,0.f};
        v8f s1 = (v8f){0.f,0.f,0.f,0.f,0.f,0.f,0.f,0.f};
#pragma unroll
        for (int kc = 0; kc < 8; ++kc) {
            const int c0 = kc * 32 + half * 8;
            v16bf b0 = pack16(*(const v8bf*)&Kt[l16][c0],
                              *(const v8bf*)&Kt[l16][c0 + 16]);
            v16bf b1 = pack16(*(const v8bf*)&Kt[16 + l16][c0],
                              *(const v8bf*)&Kt[16 + l16][c0 + 16]);
            s0 = wmma_bf16(qf[kc], b0, s0);
            s1 = wmma_bf16(qf[kc], b1, s1);
        }

        // ---- online softmax per row (row = half*8 + r, cols across l16) ----
#pragma unroll
        for (int r = 0; r < 8; ++r) {
            float a0 = s0[r] * scale;
            float a1 = s1[r] * scale;
            float mr = fmaxf(a0, a1);
#pragma unroll
            for (int off = 1; off < 16; off <<= 1)
                mr = fmaxf(mr, __shfl_xor(mr, off, 32));
            float nm = fmaxf(run_m[r], mr);
            float f  = __expf(run_m[r] - nm);
            run_m[r] = nm;
            float p0 = __expf(a0 - nm);
            float p1 = __expf(a1 - nm);
            float ps = p0 + p1;
#pragma unroll
            for (int off = 1; off < 16; off <<= 1)
                ps += __shfl_xor(ps, off, 32);
            run_l[r] = run_l[r] * f + ps;
            // store P tile (C/D layout -> LDS) for re-load as A fragment
            Pl[wave][half * 8 + r][l16]      = (__bf16)p0;
            Pl[wave][half * 8 + r][16 + l16] = (__bf16)p1;
            // rescale O accumulators for this row
#pragma unroll
            for (int cf = 0; cf < 16; ++cf) o[cf][r] *= f;
        }
        __syncthreads();

        // ---- AV: O[16q x 256c] += P(16x32) @ V^T(32x16 per c-chunk) ----
        const int pkb = half * 8;
        v16bf pf = pack16(*(const v8bf*)&Pl[wave][l16][pkb],
                          *(const v8bf*)&Pl[wave][l16][pkb + 16]);
#pragma unroll
        for (int cf = 0; cf < 16; ++cf) {
            const int cc = cf * 16 + l16;
            const int kh = half * 8;
            v16bf vb = pack16(*(const v8bf*)&Vl[cc][kh],
                              *(const v8bf*)&Vl[cc][kh + 16]);
            o[cf] = wmma_bf16(pf, vb, o[cf]);
        }
        __syncthreads();
    }

    // ---- normalize and write out av[C, N] ----
    float inv[8];
#pragma unroll
    for (int r = 0; r < 8; ++r) inv[r] = 1.f / run_l[r];
#pragma unroll
    for (int cf = 0; cf < 16; ++cf)
#pragma unroll
        for (int r = 0; r < 8; ++r) {
            int c = cf * 16 + l16;
            int n = n0 + wave * 16 + half * 8 + r;
            Oav[(size_t)b * C * N + (size_t)c * N + n] = o[cf][r] * inv[r];
        }
}

// ---------------------------------------------------------------------------
// Block-average pool all text maps to 8x8 and concat channels.
// cat: [B, 1024, 64]
// ---------------------------------------------------------------------------
__global__ void pool_concat_kernel(const float* __restrict__ t0,
                                   const float* __restrict__ t1,
                                   const float* __restrict__ t2,
                                   const float* __restrict__ t3,
                                   float* __restrict__ cat)
{
    int idx = blockIdx.x * blockDim.x + threadIdx.x;
    if (idx >= 4 * 1024 * 64) return;
    int s8 = idx & 63;
    int ch = (idx >> 6) & 1023;
    int b  = idx >> 16;
    int si = ch >> 8;
    int c  = ch & 255;
    const float* t = (si == 0) ? t0 : (si == 1) ? t1 : (si == 2) ? t2 : t3;
    int H = 64 >> si;
    int f = H >> 3;
    int y8 = s8 >> 3, x8 = s8 & 7;
    const float* base = t + ((size_t)b * 256 + c) * H * H;
    float s = 0.f;
    for (int fy = 0; fy < f; ++fy)
        for (int fx = 0; fx < f; ++fx)
            s += base[(y8 * f + fy) * H + (x8 * f + fx)];
    cat[idx] = s / (float)(f * f);
}

// ---------------------------------------------------------------------------
// BatchNorm (training mode, biased var over B,H,W) + ReLU + global avg pool.
// x: [4, 256, 64] -> pooled: [4, 256]. One block, 256 threads (1 per channel).
// ---------------------------------------------------------------------------
__global__ __launch_bounds__(256)
void bn_relu_pool_kernel(const float* __restrict__ x,
                         const float* __restrict__ g,
                         const float* __restrict__ be,
                         float* __restrict__ pooled)
{
    int c = threadIdx.x;
    float s = 0.f, s2 = 0.f;
    for (int b = 0; b < 4; ++b)
        for (int i = 0; i < 64; ++i) {
            float v = x[((size_t)b * 256 + c) * 64 + i];
            s += v; s2 += v * v;
        }
    float mu  = s * (1.f / 256.f);
    float var = s2 * (1.f / 256.f) - mu * mu;
    float rs  = rsqrtf(var + 1e-5f);
    float gg = g[c], bb = be[c];
    for (int b = 0; b < 4; ++b) {
        float a = 0.f;
        for (int i = 0; i < 64; ++i) {
            float v = (x[((size_t)b * 256 + c) * 64 + i] - mu) * rs * gg + bb;
            a += fmaxf(v, 0.f);
        }
        pooled[b * 256 + c] = a * (1.f / 64.f);
    }
}

// ---------------------------------------------------------------------------
// Tiny FC: y[b,o] = sum_i x[b,i]*w[o,i] + bias[o], optional ReLU. (8 MFLOP)
// ---------------------------------------------------------------------------
__global__ void fc_kernel(const float* __restrict__ x,
                          const float* __restrict__ w,
                          const float* __restrict__ bias,
                          float* __restrict__ y,
                          int Bn, int In, int On, int relu)
{
    int idx = blockIdx.x * blockDim.x + threadIdx.x;
    if (idx >= Bn * On) return;
    int o = idx % On, b = idx / On;
    const float* xr = x + (size_t)b * In;
    const float* wr = w + (size_t)o * In;
    float a = bias[o];
    for (int i = 0; i < In; ++i) a += xr[i] * wr[i];
    if (relu) a = fmaxf(a, 0.f);
    y[idx] = a;
}

// ---------------------------------------------------------------------------
// Host launcher
// ---------------------------------------------------------------------------
extern "C" void kernel_launch(void* const* d_in, const int* in_sizes, int n_in,
                              void* d_out, int out_size, void* d_ws, size_t ws_size,
                              hipStream_t stream)
{
    (void)in_sizes; (void)n_in; (void)out_size; (void)ws_size;

    const float* v[4]   = {(const float*)d_in[0], (const float*)d_in[1],
                           (const float*)d_in[2], (const float*)d_in[3]};
    const float* t[4]   = {(const float*)d_in[4], (const float*)d_in[5],
                           (const float*)d_in[6], (const float*)d_in[7]};
    const float* vp_w[4], *vp_b[4], *op_w[4], *op_b[4];
    for (int i = 0; i < 4; ++i) {
        vp_w[i] = (const float*)d_in[8 + 4 * i + 0];
        vp_b[i] = (const float*)d_in[8 + 4 * i + 1];
        op_w[i] = (const float*)d_in[8 + 4 * i + 2];
        op_b[i] = (const float*)d_in[8 + 4 * i + 3];
    }
    const float* q_w   = (const float*)d_in[24];
    const float* q_b   = (const float*)d_in[25];
    const float* k_w   = (const float*)d_in[26];
    const float* k_b   = (const float*)d_in[27];
    const float* vv_w  = (const float*)d_in[28];
    const float* vv_b  = (const float*)d_in[29];
    const float* sg_w  = (const float*)d_in[30];
    const float* sg_b  = (const float*)d_in[31];
    const float* bn_g  = (const float*)d_in[32];
    const float* bn_b  = (const float*)d_in[33];
    const float* fc1_w = (const float*)d_in[34];
    const float* fc1_b = (const float*)d_in[35];
    const float* fc2_w = (const float*)d_in[36];
    const float* fc2_b = (const float*)d_in[37];

    float* outp = (float*)d_out;
    float* ws   = (float*)d_ws;

    const size_t MAXE = (size_t)4 * 256 * 4096;   // 4,194,304 floats
    float* ws_vp   = ws;                 // vp, later reused as av
    float* ws_q    = ws + MAXE;
    float* ws_k    = ws + 2 * MAXE;
    float* ws_v    = ws + 3 * MAXE;
    float* ws_cat  = ws + 4 * MAXE;                  // 262,144
    float* ws_x    = ws_cat + (size_t)4 * 1024 * 64; //  65,536
    float* ws_pool = ws_x + (size_t)4 * 256 * 64;    //   1,024
    float* ws_fc1  = ws_pool + 4 * 256;              //   4,096

    const int   cin[4]      = {64, 128, 256, 512};
    const int   Ns[4]       = {4096, 1024, 256, 64};
    const size_t enh_off[4] = {8192, 1056768, 1581056, 1843200};
    const float scale = 0.0625f;   // 1/sqrt(256)

    dim3 blk(128);
    for (int i = 0; i < 4; ++i) {
        const int N = Ns[i], Ci = cin[i];
        // vp = vp_w @ v + vp_b          [B, 256, N]
        gemm64x64_bf16wmma<<<dim3(N / 64, 4, 4), blk, 0, stream>>>(
            vp_w[i], v[i], vp_b[i], nullptr, ws_vp, 256, N, Ci);
        // q = q_w @ vp + q_b
        gemm64x64_bf16wmma<<<dim3(N / 64, 4, 4), blk, 0, stream>>>(
            q_w, ws_vp, q_b, nullptr, ws_q, 256, N, 256);
        // k = k_w @ t + k_b ; v = v_w @ t + v_b
        gemm64x64_bf16wmma<<<dim3(N / 64, 4, 4), blk, 0, stream>>>(
            k_w, t[i], k_b, nullptr, ws_k, 256, N, 256);
        gemm64x64_bf16wmma<<<dim3(N / 64, 4, 4), blk, 0, stream>>>(
            vv_w, t[i], vv_b, nullptr, ws_v, 256, N, 256);
        // av = flash_attn(q, k, v)  -> overwrite ws_vp
        attn_flash_bf16wmma<<<dim3(N / 64, 1, 4), blk, 0, stream>>>(
            ws_q, ws_k, ws_v, ws_vp, N, scale);
        // enhanced = op_w @ av + op_b + v_i
        gemm64x64_bf16wmma<<<dim3(N / 64, Ci / 64, 4), blk, 0, stream>>>(
            op_w[i], ws_vp, op_b[i], v[i], outp + enh_off[i], Ci, N, 256);
    }

    // semantic branch
    pool_concat_kernel<<<(4 * 1024 * 64 + 255) / 256, 256, 0, stream>>>(
        t[0], t[1], t[2], t[3], ws_cat);
    gemm64x64_bf16wmma<<<dim3(1, 4, 4), blk, 0, stream>>>(
        sg_w, ws_cat, sg_b, nullptr, ws_x, 256, 64, 1024);
    bn_relu_pool_kernel<<<1, 256, 0, stream>>>(ws_x, bn_g, bn_b, ws_pool);
    fc_kernel<<<(4 * 1024 + 255) / 256, 256, 0, stream>>>(
        ws_pool, fc1_w, fc1_b, ws_fc1, 4, 256, 1024, 1);
    fc_kernel<<<(4 * 2048 + 255) / 256, 256, 0, stream>>>(
        ws_fc1, fc2_w, fc2_b, outp, 4, 1024, 2048, 0);
}
